// GAT_23630910063029
// MI455X (gfx1250) — matrix operand order repaired and checked by reading, not compiled, not run on hardware
//
#include <hip/hip_runtime.h>
#include <math.h>

#define Nn   100000
#define Ee   1600000
#define ENp  (Ee + Nn)        // edges + self loops
#define DIN  128
#define Ff   64
#define F3   (3 * Ff)
#define Gg   512
#define Cc   10

typedef __attribute__((ext_vector_type(16))) _Float16 v16h;
typedef __attribute__((ext_vector_type(8)))  float    v8f;

// ---------------------------------------------------------------- utilities
__global__ void k_zero(float* __restrict__ p, long n) {
    long i = (long)blockIdx.x * blockDim.x + threadIdx.x;
    long stride = (long)gridDim.x * blockDim.x;
    for (; i < n; i += stride) p[i] = 0.f;
}

__device__ __forceinline__ void atomicMaxF(float* addr, float v) {
    // ordered int trick; init value is a large negative float
    if (v >= 0.f) atomicMax((int*)addr, __float_as_int(v));
    else          atomicMin((unsigned int*)addr, __float_as_uint(v));
}

// ------------------------------------------------- W pre-pack (f16 B frags)
// Packed layout: entry t = (chunk*2 + kh)*64 + n holds 16 halfs j=0..15 with
// K = chunk*32 + kh*16 + j, column n.  One 32B vector per lane in the GEMM.
__global__ void k_pack_w(const float* __restrict__ W, int K, _Float16* __restrict__ Wp) {
    int t = threadIdx.x;                       // single block, 512 threads
    int entries = (K / 32) * 2 * 64;
    if (t >= entries) return;
    int c  = t >> 7;                           // chunk
    int kh = (t >> 6) & 1;
    int n  = t & 63;
    _Float16* dst = Wp + (size_t)t * 16;
    #pragma unroll
    for (int j = 0; j < 16; j++)
        dst[j] = (_Float16)W[(size_t)(c * 32 + kh * 16 + j) * Ff + n];
}

// ---------------------------------------------------------------- WMMA GEMM
// H[N,64] = A[N,K] (row stride lda, fp32 -> f16 on the fly) * W (pre-packed f16)
// grid.x = N/32 (2 row tiles per block), block = 128 (4 waves = 4 col tiles).
// K is a compile-time constant -> fully unrolled K loop, B frags in registers.
template <int K>
__global__ __launch_bounds__(128)
void k_gemm_wmma(const float* __restrict__ A, int lda,
                 const _Float16* __restrict__ Wp, float* __restrict__ H) {
    constexpr int NC = K / 32;                 // # of 32-wide K chunks
    const int wave = threadIdx.x >> 5;         // col tile 0..3
    const int lane = threadIdx.x & 31;
    const int ml   = lane & 15;
    const int kh   = lane >> 4;                // 0 or 1
    const int col  = wave * 16 + ml;           // W col for this lane (B frag)
    const int rowTile0 = blockIdx.x * 2;

    // preload all B fragments for this column tile (<= 4 x v16h)
    v16h bfrag[NC];
    #pragma unroll
    for (int c = 0; c < NC; c++)
        bfrag[c] = *(const v16h*)(Wp + (size_t)((c * 2 + kh) * 64 + col) * 16);

    #pragma unroll
    for (int r = 0; r < 2; r++) {
        const int row = (rowTile0 + r) * 16 + ml;
        v8f acc = {0.f, 0.f, 0.f, 0.f, 0.f, 0.f, 0.f, 0.f};

        #pragma unroll
        for (int c = 0; c < NC; c++) {
            // A fragment: contiguous k order, same as packed W.
            // half kh: VGPR0-7 hold K = kh*16+0..15
            union { v16h v; _Float16 h[16]; } a;
            const float* ap = A + (size_t)row * lda + c * 32 + kh * 16;
            float4 x0 = *(const float4*)(ap + 0);
            float4 x1 = *(const float4*)(ap + 4);
            float4 y0 = *(const float4*)(ap + 8);
            float4 y1 = *(const float4*)(ap + 12);
            a.h[0]  = (_Float16)x0.x; a.h[1]  = (_Float16)x0.y;
            a.h[2]  = (_Float16)x0.z; a.h[3]  = (_Float16)x0.w;
            a.h[4]  = (_Float16)x1.x; a.h[5]  = (_Float16)x1.y;
            a.h[6]  = (_Float16)x1.z; a.h[7]  = (_Float16)x1.w;
            a.h[8]  = (_Float16)y0.x; a.h[9]  = (_Float16)y0.y;
            a.h[10] = (_Float16)y0.z; a.h[11] = (_Float16)y0.w;
            a.h[12] = (_Float16)y1.x; a.h[13] = (_Float16)y1.y;
            a.h[14] = (_Float16)y1.z; a.h[15] = (_Float16)y1.w;

            acc = __builtin_amdgcn_wmma_f32_16x16x32_f16(
                    /*neg_a=*/false, a.v, /*neg_b=*/false, bfrag[c],
                    /*c_mod=*/(short)0, acc, /*reuse_a=*/false, /*reuse_b=*/false);
        }

        // C/D layout: lane l -> n = l&15, rows m = (l>>4)*8 + rr
        union { v8f v; float f[8]; } cu; cu.v = acc;
        float* hp = H + (size_t)((rowTile0 + r) * 16 + kh * 8) * Ff + wave * 16 + ml;
        #pragma unroll
        for (int rr = 0; rr < 8; rr++) hp[(size_t)rr * Ff] = cu.f[rr];
    }
}

// ------------------------------------------------------------- node scoring
__global__ void k_node_prep(const float* __restrict__ h,
                            const float* __restrict__ as_, const float* __restrict__ ad_,
                            float* __restrict__ esrc, float* __restrict__ edst,
                            float* __restrict__ mmax, float* __restrict__ denom) {
    int i = blockIdx.x * blockDim.x + threadIdx.x;
    if (i >= Nn) return;
    const float* hp = h + (size_t)i * Ff;
    float s1 = 0.f, s2 = 0.f;
    #pragma unroll 8
    for (int f = 0; f < Ff; f++) { float v = hp[f]; s1 += v * as_[f]; s2 += v * ad_[f]; }
    esrc[i] = s1; edst[i] = s2;
    mmax[i] = -3.0e38f; denom[i] = 0.f;
}

// ------------------------------------------------------------- edge passes
__device__ __forceinline__ void edge_sd(const int* __restrict__ ei, int i, int& s, int& d) {
    if (i < Ee) { s = ei[i]; d = ei[Ee + i]; } else { s = d = i - Ee; }
}

// pass 1: leaky-relu score -> ew, running segment max
__global__ void k_edge_score_max(const int* __restrict__ ei,
                                 const float* __restrict__ esrc, const float* __restrict__ edst,
                                 float* __restrict__ mmax, float* __restrict__ ew) {
    int i = blockIdx.x * blockDim.x + threadIdx.x;
    if (i >= ENp) return;
    int s, d; edge_sd(ei, i, s, d);
    float e = esrc[s] + edst[d];
    e = e > 0.f ? e : 0.2f * e;                 // leaky relu
    ew[i] = e;
    atomicMaxF(&mmax[d], e);
}

// pass 2: exp(e - max) in place, running segment sum (dst index only)
__global__ void k_edge_expsum(const int* __restrict__ ei,
                              const float* __restrict__ mmax, float* __restrict__ denom,
                              float* __restrict__ ew) {
    int i = blockIdx.x * blockDim.x + threadIdx.x;
    if (i >= ENp) return;
    int d = (i < Ee) ? ei[Ee + i] : (i - Ee);
    float w = __expf(ew[i] - mmax[d]);
    ew[i] = w;
    atomicAdd(&denom[d], w);
}

// pass 3: 16 lanes per edge, 4 features per lane (float4 gather + fp32 atomics)
__global__ void k_edge_aggregate(const int* __restrict__ ei,
                                 const float* __restrict__ ew, const float* __restrict__ denom,
                                 const float* __restrict__ h, float* __restrict__ out) {
    int t = blockIdx.x * blockDim.x + threadIdx.x;
    int edge = t >> 4;
    int fc = (t & 15) * 4;
    if (edge >= ENp) return;
    int s, d; edge_sd(ei, edge, s, d);
    float alpha = ew[edge] / (denom[d] + 1e-16f);
    float4 hv = *(const float4*)(h + (size_t)s * Ff + fc);
    float* op = out + (size_t)d * Ff + fc;
    atomicAdd(op + 0, alpha * hv.x);
    atomicAdd(op + 1, alpha * hv.y);
    atomicAdd(op + 2, alpha * hv.z);
    atomicAdd(op + 3, alpha * hv.w);
}

// -------------------------------------------------- bias + relu into concat
__global__ void k_bias_relu_concat(const float* __restrict__ out, const float* __restrict__ bias,
                                   float* __restrict__ xc, int off) {
    long i = (long)blockIdx.x * blockDim.x + threadIdx.x;
    if (i >= (long)Nn * Ff) return;
    int node = (int)(i >> 6);
    int f = (int)(i & 63);
    float v = out[i] + bias[f];
    v = v > 0.f ? v : 0.f;
    xc[(size_t)node * F3 + off + f] = v;
}

// ---------------------------------------------------------------- pooling
__global__ void k_pool(const float* __restrict__ xc, const int* __restrict__ batch,
                       float* __restrict__ gsum, float* __restrict__ gcnt) {
    int i = blockIdx.x * blockDim.x + threadIdx.x;
    if (i >= Nn) return;
    int g = batch[i];
    const float* xp = xc + (size_t)i * F3;
    float* gp = gsum + (size_t)g * F3;
    for (int f = 0; f < F3; f += 4) {
        float4 v = *(const float4*)(xp + f);
        atomicAdd(gp + f + 0, v.x);
        atomicAdd(gp + f + 1, v.y);
        atomicAdd(gp + f + 2, v.z);
        atomicAdd(gp + f + 3, v.w);
    }
    atomicAdd(&gcnt[g], 1.f);
}

// ---------------------------------------------------- classifier + softmax
__global__ void k_final(const float* __restrict__ gsum, const float* __restrict__ gcnt,
                        const float* __restrict__ Wo, const float* __restrict__ bo,
                        float* __restrict__ outp) {
    int g = blockIdx.x * blockDim.x + threadIdx.x;
    if (g >= Gg) return;
    float inv = 1.f / fmaxf(gcnt[g], 1.f);
    float acc[Cc];
    #pragma unroll
    for (int c = 0; c < Cc; c++) acc[c] = bo[c];
    const float* gp = gsum + (size_t)g * F3;
    for (int f = 0; f < F3; f++) {
        float p = gp[f] * inv;
        #pragma unroll
        for (int c = 0; c < Cc; c++) acc[c] += p * Wo[f * Cc + c];
    }
    float mx = acc[0];
    #pragma unroll
    for (int c = 1; c < Cc; c++) mx = fmaxf(mx, acc[c]);
    float s = 0.f;
    #pragma unroll
    for (int c = 0; c < Cc; c++) { acc[c] = __expf(acc[c] - mx); s += acc[c]; }
    float is = 1.f / s;
    #pragma unroll
    for (int c = 0; c < Cc; c++) outp[(size_t)g * Cc + c] = acc[c] * is;
}

// ---------------------------------------------------------------- driver
template <int K>
static void run_layer(const float* Ain, int lda,
                      const float* Wm, const float* as_, const float* ad_, const float* bias,
                      const int* ei,
                      float* h, float* out, float* esrc, float* edst,
                      float* mmax, float* denom, float* ew, _Float16* wpack,
                      float* xc, int off, hipStream_t stream) {
    k_pack_w<<<1, 512, 0, stream>>>(Wm, K, wpack);
    k_gemm_wmma<K><<<Nn / 32, 128, 0, stream>>>(Ain, lda, wpack, h);
    k_zero<<<2048, 256, 0, stream>>>(out, (long)Nn * Ff);
    k_node_prep<<<(Nn + 255) / 256, 256, 0, stream>>>(h, as_, ad_, esrc, edst, mmax, denom);
    k_edge_score_max<<<(ENp + 255) / 256, 256, 0, stream>>>(ei, esrc, edst, mmax, ew);
    k_edge_expsum<<<(ENp + 255) / 256, 256, 0, stream>>>(ei, mmax, denom, ew);
    k_edge_aggregate<<<(ENp + 15) / 16, 256, 0, stream>>>(ei, ew, denom, h, out);
    k_bias_relu_concat<<<((long)Nn * Ff + 255) / 256, 256, 0, stream>>>(out, bias, xc, off);
}

extern "C" void kernel_launch(void* const* d_in, const int* in_sizes, int n_in,
                              void* d_out, int out_size, void* d_ws, size_t ws_size,
                              hipStream_t stream) {
    (void)in_sizes; (void)n_in; (void)out_size; (void)ws_size;

    const float* x     = (const float*)d_in[0];
    const int*   ei    = (const int*)d_in[1];
    const int*   batch = (const int*)d_in[2];
    const float* W1 = (const float*)d_in[3];
    const float* a1s = (const float*)d_in[4];
    const float* a1d = (const float*)d_in[5];
    const float* b1 = (const float*)d_in[6];
    const float* W2 = (const float*)d_in[7];
    const float* a2s = (const float*)d_in[8];
    const float* a2d = (const float*)d_in[9];
    const float* b2 = (const float*)d_in[10];
    const float* W3 = (const float*)d_in[11];
    const float* a3s = (const float*)d_in[12];
    const float* a3d = (const float*)d_in[13];
    const float* b3 = (const float*)d_in[14];
    const float* Wo = (const float*)d_in[15];
    const float* bo = (const float*)d_in[16];

    // workspace layout (floats): ~137 MB total
    float* ws    = (float*)d_ws;
    float* h     = ws;                                 // N*64
    float* out   = h     + (size_t)Nn * Ff;            // N*64
    float* xc    = out   + (size_t)Nn * Ff;            // N*192
    float* ew    = xc    + (size_t)Nn * F3;            // E+N
    float* esrc  = ew    + (size_t)ENp;                // N
    float* edst  = esrc  + Nn;                         // N
    float* mmax  = edst  + Nn;                         // N
    float* denom = mmax  + Nn;                         // N
    float* gsum  = denom + Nn;                         // G*192
    float* gcnt  = gsum  + (size_t)Gg * F3;            // G (contiguous after gsum)
    _Float16* wpack = (_Float16*)(gcnt + Gg);          // <= 8192 halfs (16 KB)

    // layer 1: input x [N,128]
    run_layer<DIN>(x, DIN, W1, a1s, a1d, b1, ei,
                   h, out, esrc, edst, mmax, denom, ew, wpack, xc, 0, stream);
    // layer 2: input xc[:,0:64]
    run_layer<Ff>(xc, F3, W2, a2s, a2d, b2, ei,
                  h, out, esrc, edst, mmax, denom, ew, wpack, xc, Ff, stream);
    // layer 3: input xc[:,64:128]
    run_layer<Ff>(xc + Ff, F3, W3, a3s, a3d, b3, ei,
                  h, out, esrc, edst, mmax, denom, ew, wpack, xc, 2 * Ff, stream);

    // mean pooling + classifier + softmax
    k_zero<<<256, 256, 0, stream>>>(gsum, (long)Gg * F3 + Gg);
    k_pool<<<(Nn + 255) / 256, 256, 0, stream>>>(xc, batch, gsum, gcnt);
    k_final<<<(Gg + 255) / 256, 256, 0, stream>>>(gsum, gcnt, Wo, bo, (float*)d_out);
}
